// LSTMFinder_28243704938952
// MI455X (gfx1250) — matrix operand order repaired and checked by reading, not compiled, not run on hardware
//
#include <hip/hip_runtime.h>
#include <hip/hip_bf16.h>

// ---------------------------------------------------------------------------
// Problem constants (from the reference)
// ---------------------------------------------------------------------------
#define EMB   512          // E
#define UNITS 1024         // H = 2E
#define BATCH 512          // B
#define KCAND 32           // K candidates
#define TSTEP 8            // T
#define FOURH 4096         // 4H
#define KZ    2048         // K-dim of fused LSTM gemm ([x,h])
#define MLPK  2048         // 4E   (ent | h | relation)
#define N1    1536         // 3E
#define N2    1024         // 2E

typedef __attribute__((ext_vector_type(16))) __bf16       v16bf;
typedef __attribute__((ext_vector_type(8)))  float        v8f;
typedef __attribute__((ext_vector_type(4)))  unsigned int v4u;

// GCC-style 4 x i32 vector: the pointee type the async-LDS builtin expects
typedef int v4i_vs __attribute__((vector_size(16)));
typedef __attribute__((address_space(1))) v4i_vs* as1_v4i_ptr;
typedef __attribute__((address_space(3))) v4i_vs* as3_v4i_ptr;

union Frag { v16bf v; v4u q[2]; };

__device__ __forceinline__ unsigned short f2bf(float x) {
    union { float f; unsigned int u; } v; v.f = x;
    unsigned int u = v.u;
    u += 0x7fffu + ((u >> 16) & 1u);           // round-to-nearest-even
    return (unsigned short)(u >> 16);
}
__device__ __forceinline__ float bf2f(unsigned short b) {
    union { unsigned int u; float f; } v; v.u = ((unsigned int)b) << 16;
    return v.f;
}
__device__ __forceinline__ float sigmoidf(float x) {
    return 1.0f / (1.0f + expf(-x));
}

// ---------------------------------------------------------------------------
// CDNA5 async global->LDS copy (ASYNCcnt) with fallbacks
// ---------------------------------------------------------------------------
__device__ __forceinline__ void async_b128(unsigned short* lds, const unsigned short* g) {
#if __has_builtin(__builtin_amdgcn_global_load_async_to_lds_b128)
    __builtin_amdgcn_global_load_async_to_lds_b128(
        (as1_v4i_ptr)(unsigned long long)g,
        (as3_v4i_ptr)(unsigned int)(unsigned long long)lds,
        0, 0);
#else
    unsigned int l = (unsigned int)(unsigned long long)lds;   // LDS offset = addr[31:0]
    asm volatile("global_load_async_to_lds_b128 %0, %1, off"
                 :: "v"(l), "v"(g) : "memory");
#endif
}

__device__ __forceinline__ void wait_async0() {
#if __has_builtin(__builtin_amdgcn_s_wait_asynccnt)
    __builtin_amdgcn_s_wait_asynccnt(0);
#else
    asm volatile("s_wait_asynccnt 0" ::: "memory");
#endif
}

// ---------------------------------------------------------------------------
// Setup kernels
// ---------------------------------------------------------------------------
// Swizzle f32 weights [K,N] row-major -> bf16 fragment-ready tiles:
//   dst[((ntile*(K/32)+kt)*4096) + n*32 + kk] = bf16(src[(kt*32+kk)*N + ntile*128 + n])
__global__ void k_swz(const float* __restrict__ src, unsigned short* __restrict__ dst,
                      int N, int K) {
    int idx = blockIdx.x * blockDim.x + threadIdx.x;
    if (idx >= N * K) return;
    int t   = idx >> 12;            // /4096
    int rkk = idx & 4095;
    int n   = rkk >> 5, kk = rkk & 31;
    int nkt = K >> 5;
    int ntile = t / nkt, kt = t - ntile * nkt;
    int ncol = ntile * 128 + n, k = kt * 32 + kk;
    dst[idx] = f2bf(src[(size_t)k * N + ncol]);
}

// Same, for the fused Wz = [Wx ; Wh]  (K=2048, N=4096)
__global__ void k_swz_wz(const float* __restrict__ Wx, const float* __restrict__ Wh,
                         unsigned short* __restrict__ dst) {
    int idx = blockIdx.x * blockDim.x + threadIdx.x;
    if (idx >= KZ * FOURH) return;
    int t   = idx >> 12;
    int rkk = idx & 4095;
    int n   = rkk >> 5, kk = rkk & 31;
    int nkt = KZ >> 5;                       // 64
    int ntile = t / nkt, kt = t - ntile * nkt;
    int ncol = ntile * 128 + n, k = kt * 32 + kk;
    float v = (k < UNITS) ? Wx[(size_t)k * FOURH + ncol]
                          : Wh[(size_t)(k - UNITS) * FOURH + ncol];
    dst[idx] = f2bf(v);
}

// x_all[T*B, 1024] bf16 = [rel_emb[prl] | ent_emb[cur]]
__global__ void k_gather_x(const float* __restrict__ rel_emb, const float* __restrict__ ent_emb,
                           const int* __restrict__ prl, const int* __restrict__ cur,
                           unsigned short* __restrict__ x_all) {
    int idx = blockIdx.x * blockDim.x + threadIdx.x;
    if (idx >= TSTEP * BATCH * UNITS) return;
    int row = idx / UNITS, j = idx - row * UNITS;
    float v = (j < EMB) ? rel_emb[(size_t)prl[row] * EMB + j]
                        : ent_emb[(size_t)cur[row] * EMB + (j - EMB)];
    x_all[idx] = f2bf(v);
}

// MLPin[:, 1536:2048] = bf16(relation)   (constant over steps)
__global__ void k_fill_relation(const float* __restrict__ relation,
                                unsigned short* __restrict__ mlp_in) {
    int idx = blockIdx.x * blockDim.x + threadIdx.x;
    if (idx >= BATCH * EMB) return;
    int b = idx / EMB, j = idx - b * EMB;
    mlp_in[b * MLPK + N1 + j] = f2bf(relation[idx]);
}

// h part of XH := 0, c := 0
__global__ void k_init_state(unsigned short* __restrict__ xh, float* __restrict__ c) {
    int idx = blockIdx.x * blockDim.x + threadIdx.x;
    if (idx >= BATCH * UNITS) return;
    int b = idx / UNITS, j = idx - b * UNITS;
    xh[b * KZ + UNITS + j] = 0;
    c[idx] = 0.0f;
}

// Per step: XH[:, 0:1024] = x_t ; MLPin[:, 0:512] = bf16(ent_emb[cur_t])
__global__ void k_prep_step(const unsigned short* __restrict__ x_t,
                            const float* __restrict__ ent_emb,
                            const int* __restrict__ cur_t,
                            unsigned short* __restrict__ xh,
                            unsigned short* __restrict__ mlp_in) {
    int idx = blockIdx.x * blockDim.x + threadIdx.x;
    if (idx >= BATCH * UNITS) return;
    int b = idx / UNITS, j = idx - b * UNITS;
    xh[b * KZ + j] = x_t[idx];
    if (j < EMB)
        mlp_in[b * MLPK + j] = f2bf(ent_emb[(size_t)cur_t[b] * EMB + j]);
}

// ---------------------------------------------------------------------------
// WMMA GEMM:  C[512,N] = A[512,K](bf16,row) * Bsw(swizzled bf16) + bias
//   EPI 0 : C -> f32      EPI 1 : C -> relu -> bf16
// 256 threads, block tile 128x128, wave tile 64x32, K step 32,
// double-buffered LDS fed by async global->LDS B128 copies.
// ---------------------------------------------------------------------------
template <int N, int KDIM, int EPI>
__global__ __launch_bounds__(256) void k_gemm(
        const unsigned short* __restrict__ A,
        const unsigned short* __restrict__ Bsw,
        const float* __restrict__ bias,
        void* __restrict__ Cout)
{
    __shared__ __align__(16) unsigned short As[2][128][40];
    __shared__ __align__(16) unsigned short Bs[2][128][40];
    constexpr int NKT = KDIM / 32;

    const int tid  = threadIdx.x;
    const int lane = tid & 31;
    const int wave = tid >> 5;
    const int wm   = (wave >> 2) * 64;
    const int wn   = (wave & 3) * 32;
    const int m0   = blockIdx.y * 128;
    const int n0   = blockIdx.x * 128;
    const int mrow = lane & 15;
    const int kbA  = (lane >> 4) * 8;     // A frag: lane-hi -> K+8 (ISA 16b A layout)
    const int kbB  = (lane >> 4) * 8;     // B frag: lane-hi -> K+8 (same as A)

    // per-thread staging coordinates: 512 chunks of 8 ushorts per tile
    const int r0 = tid >> 2;              // row/n for chunk tid
    const int c0 = (tid & 3) * 8;         // k offset

    auto issue = [&](int buf, int kt) {
        const unsigned short* ga = A + (size_t)(m0 + r0) * KDIM + kt * 32 + c0;
        const unsigned short* gb = Bsw + ((size_t)(blockIdx.x * NKT + kt) << 12) + tid * 8;
        async_b128(&As[buf][r0][c0], ga);                                 // chunk tid
        async_b128(&As[buf][r0 + 64][c0], ga + (size_t)64 * KDIM);        // chunk tid+256
        async_b128(&Bs[buf][r0][c0], gb);
        async_b128(&Bs[buf][r0 + 64][c0], gb + 2048);
    };

    v8f acc[4][2];
    #pragma unroll
    for (int i = 0; i < 4; ++i)
        #pragma unroll
        for (int j = 0; j < 2; ++j)
            acc[i][j] = (v8f){0.f,0.f,0.f,0.f,0.f,0.f,0.f,0.f};

    issue(0, 0);
    int cur = 0;
    for (int kt = 0; kt < NKT; ++kt) {
        wait_async0();          // own async copies for buf[cur] complete
        __syncthreads();        // everyone's copies complete; prev compute done
        if (kt + 1 < NKT) issue(cur ^ 1, kt + 1);

        Frag fa[4], fb[2];
        #pragma unroll
        for (int i = 0; i < 4; ++i) {
            const unsigned short* p = &As[cur][wm + 16 * i + mrow][kbA];
            fa[i].q[0] = *(const v4u*)p;          // K = kbA .. kbA+7
            fa[i].q[1] = *(const v4u*)(p + 16);   // K = kbA+16 .. kbA+23
        }
        #pragma unroll
        for (int j = 0; j < 2; ++j) {
            const unsigned short* p = &Bs[cur][wn + 16 * j + mrow][kbB];
            fb[j].q[0] = *(const v4u*)p;          // K = kbB .. kbB+7
            fb[j].q[1] = *(const v4u*)(p + 16);   // K = kbB+16 .. kbB+23
        }
        #pragma unroll
        for (int i = 0; i < 4; ++i)
            #pragma unroll
            for (int j = 0; j < 2; ++j)
                acc[i][j] = __builtin_amdgcn_wmma_f32_16x16x32_bf16(
                    false, fa[i].v, false, fb[j].v, (short)0, acc[i][j], false, false);
        cur ^= 1;
    }

    // epilogue: 32-bit C/D layout: VGPR r -> M = r + (lane>=16)*8, N = lane&15
    #pragma unroll
    for (int j = 0; j < 2; ++j) {
        const int ncol = n0 + wn + 16 * j + (lane & 15);
        const float bv = bias[ncol];
        #pragma unroll
        for (int i = 0; i < 4; ++i) {
            const int rbase = m0 + wm + 16 * i + (lane >> 4) * 8;
            #pragma unroll
            for (int r = 0; r < 8; ++r) {
                float v = acc[i][j][r] + bv;
                if constexpr (EPI == 1) {
                    ((unsigned short*)Cout)[(size_t)(rbase + r) * N + ncol] =
                        f2bf(fmaxf(v, 0.0f));
                } else {
                    ((float*)Cout)[(size_t)(rbase + r) * N + ncol] = v;
                }
            }
        }
    }
}

// ---------------------------------------------------------------------------
// Pointwise LSTM
// ---------------------------------------------------------------------------
__global__ void k_lstm(const float* __restrict__ z, float* __restrict__ c,
                       unsigned short* __restrict__ xh, unsigned short* __restrict__ mlp_in) {
    int idx = blockIdx.x * blockDim.x + threadIdx.x;
    if (idx >= BATCH * UNITS) return;
    int b = idx / UNITS, j = idx - b * UNITS;
    const float* zb = z + (size_t)b * FOURH;
    float i_ = zb[j];
    float f_ = zb[UNITS + j];
    float g_ = zb[2 * UNITS + j];
    float o_ = zb[3 * UNITS + j];
    float cn = sigmoidf(f_) * c[idx] + sigmoidf(i_) * tanhf(g_);
    float h  = sigmoidf(o_) * tanhf(cn);
    c[idx] = cn;
    unsigned short hb = f2bf(h);
    xh[b * KZ + UNITS + j]     = hb;
    mlp_in[b * MLPK + EMB + j] = hb;
}

// ---------------------------------------------------------------------------
// Candidate scoring + softmax. 1 block / batch row, 32 waves (1 per candidate).
// ---------------------------------------------------------------------------
__global__ __launch_bounds__(1024) void k_score_softmax(
        const float* __restrict__ ent_emb, const float* __restrict__ rel_emb,
        const unsigned short* __restrict__ feat2,
        const int* __restrict__ crel_t, const int* __restrict__ cent_t,
        float* __restrict__ out_t) {
    __shared__ float logits[KCAND];
    int b    = blockIdx.x;
    int k    = threadIdx.x >> 5;
    int lane = threadIdx.x & 31;

    const float* rp = rel_emb + (size_t)crel_t[b * KCAND + k] * EMB;
    const float* ep = ent_emb + (size_t)cent_t[b * KCAND + k] * EMB;
    const unsigned short* f = feat2 + (size_t)b * N2;

    float s = 0.0f;
    for (int d = lane; d < EMB; d += 32)
        s += rp[d] * bf2f(f[d]) + ep[d] * bf2f(f[EMB + d]);
    #pragma unroll
    for (int off = 16; off; off >>= 1) s += __shfl_xor(s, off, 32);
    if (lane == 0) logits[k] = s;
    __syncthreads();

    if (threadIdx.x < KCAND) {
        float l = logits[threadIdx.x];
        float m = l;
        #pragma unroll
        for (int off = 16; off; off >>= 1) m = fmaxf(m, __shfl_xor(m, off, 32));
        float e = expf(l - m);
        float sum = e;
        #pragma unroll
        for (int off = 16; off; off >>= 1) sum += __shfl_xor(sum, off, 32);
        out_t[b * KCAND + threadIdx.x] = e / sum;
    }
}

// ---------------------------------------------------------------------------
// Host launcher
// ---------------------------------------------------------------------------
static inline size_t align256(size_t x) { return (x + 255) & ~(size_t)255; }

extern "C" void kernel_launch(void* const* d_in, const int* in_sizes, int n_in,
                              void* d_out, int out_size, void* d_ws, size_t ws_size,
                              hipStream_t stream) {
    const float* ent_emb  = (const float*)d_in[0];
    const float* rel_emb  = (const float*)d_in[1];
    const float* relation = (const float*)d_in[2];
    const float* Wx  = (const float*)d_in[3];
    const float* Wh  = (const float*)d_in[4];
    const float* bz  = (const float*)d_in[5];
    const float* W1  = (const float*)d_in[6];
    const float* b1  = (const float*)d_in[7];
    const float* W2  = (const float*)d_in[8];
    const float* b2  = (const float*)d_in[9];
    const int* current_ids  = (const int*)d_in[10];
    const int* prev_rel_ids = (const int*)d_in[11];
    const int* cand_rel_ids = (const int*)d_in[12];
    const int* cand_ent_ids = (const int*)d_in[13];
    float* out = (float*)d_out;

    // workspace carve-up
    char* ws = (char*)d_ws;
    size_t off = 0;
    unsigned short* Wz_sw   = (unsigned short*)(ws + off); off = align256(off + (size_t)KZ * FOURH * 2);
    unsigned short* W1_sw   = (unsigned short*)(ws + off); off = align256(off + (size_t)MLPK * N1 * 2);
    unsigned short* W2_sw   = (unsigned short*)(ws + off); off = align256(off + (size_t)N1 * N2 * 2);
    unsigned short* x_all   = (unsigned short*)(ws + off); off = align256(off + (size_t)TSTEP * BATCH * UNITS * 2);
    unsigned short* XH      = (unsigned short*)(ws + off); off = align256(off + (size_t)BATCH * KZ * 2);
    float*          zbuf    = (float*)(ws + off);          off = align256(off + (size_t)BATCH * FOURH * 4);
    float*          cbuf    = (float*)(ws + off);          off = align256(off + (size_t)BATCH * UNITS * 4);
    unsigned short* MLPin   = (unsigned short*)(ws + off); off = align256(off + (size_t)BATCH * MLPK * 2);
    unsigned short* feat1   = (unsigned short*)(ws + off); off = align256(off + (size_t)BATCH * N1 * 2);
    unsigned short* feat2   = (unsigned short*)(ws + off); off = align256(off + (size_t)BATCH * N2 * 2);
    (void)off; (void)ws_size; (void)in_sizes; (void)n_in; (void)out_size;

    const int TB = 256;
    // -- per-launch setup (weights -> bf16 fragment-ready swizzle) --
    k_swz_wz<<<(KZ * FOURH + TB - 1) / TB, TB, 0, stream>>>(Wx, Wh, Wz_sw);
    k_swz<<<(MLPK * N1 + TB - 1) / TB, TB, 0, stream>>>(W1, W1_sw, N1, MLPK);
    k_swz<<<(N1 * N2 + TB - 1) / TB, TB, 0, stream>>>(W2, W2_sw, N2, N1);
    k_gather_x<<<(TSTEP * BATCH * UNITS + TB - 1) / TB, TB, 0, stream>>>(
        rel_emb, ent_emb, prev_rel_ids, current_ids, x_all);
    k_fill_relation<<<(BATCH * EMB + TB - 1) / TB, TB, 0, stream>>>(relation, MLPin);
    k_init_state<<<(BATCH * UNITS + TB - 1) / TB, TB, 0, stream>>>(XH, cbuf);

    // -- sequential timesteps --
    for (int t = 0; t < TSTEP; ++t) {
        const unsigned short* x_t = x_all + (size_t)t * BATCH * UNITS;
        const int* cur_t  = current_ids  + t * BATCH;
        const int* crel_t = cand_rel_ids + (size_t)t * BATCH * KCAND;
        const int* cent_t = cand_ent_ids + (size_t)t * BATCH * KCAND;
        float* out_t = out + (size_t)t * BATCH * KCAND;

        k_prep_step<<<(BATCH * UNITS + TB - 1) / TB, TB, 0, stream>>>(
            x_t, ent_emb, cur_t, XH, MLPin);

        // z = [x,h] @ Wz + b    (M=512, N=4096, K=2048) -> f32
        k_gemm<FOURH, KZ, 0><<<dim3(FOURH / 128, BATCH / 128), 256, 0, stream>>>(
            XH, Wz_sw, bz, zbuf);

        k_lstm<<<(BATCH * UNITS + TB - 1) / TB, TB, 0, stream>>>(zbuf, cbuf, XH, MLPin);

        // feat1 = relu([ent,h,rel] @ W1 + b1)  (M=512, N=1536, K=2048) -> bf16
        k_gemm<N1, MLPK, 1><<<dim3(N1 / 128, BATCH / 128), 256, 0, stream>>>(
            MLPin, W1_sw, b1, feat1);

        // feat2 = relu(feat1 @ W2 + b2)        (M=512, N=1024, K=1536) -> bf16
        k_gemm<N2, N1, 1><<<dim3(N2 / 128, BATCH / 128), 256, 0, stream>>>(
            feat1, W2_sw, b2, feat2);

        k_score_softmax<<<BATCH, 1024, 0, stream>>>(
            ent_emb, rel_emb, feat2, crel_t, cent_t, out_t);
    }
}
